// MultiHeadSelfAttention_21569325760919
// MI455X (gfx1250) — matrix operand order repaired and checked by reading, not compiled, not run on hardware
//
#include <hip/hip_runtime.h>
#include <hip/hip_bf16.h>

// ---------------------------------------------------------------------------
// Multi-head self-attention for MI455X (gfx1250, wave32, WMMA bf16 path)
// Pipeline:
//   0) f32->bf16 pre-conversion of X, W_qkv, W_o (bandwidth-trivial)
//   1) gemm<3072,QKV> : Xbf @ Wqkv_bf + b -> Q/K/V bf16 [B,H,N,64]
//      (double-buffered LDS, async global->LDS A-panel staging)
//   2) attn : flash-attention per (b,h), WMMA for QK^T and PV
//   3) gemm<1024,OUT> : attn_out(bf16) @ Wo_bf + b_o -> f32 out
// ---------------------------------------------------------------------------

typedef __attribute__((ext_vector_type(16))) __bf16 v16bf;
typedef __attribute__((ext_vector_type(8)))  __bf16 v8bf;
typedef __attribute__((ext_vector_type(8)))  float  v8f;
typedef __attribute__((ext_vector_type(8)))  unsigned short v8u;
typedef __attribute__((ext_vector_type(4)))  unsigned short v4u;
typedef int b128_t __attribute__((vector_size(16)));   // matches builtin param

#define D_MODEL   1024
#define NUM_HEADS 16
#define D_K       64
#define BATCH     8
#define SEQ       1024

#define BM 128
#define BN 128
#define BK 32
#define LDA 40   // LDS row stride in bf16 elements (80B, 16B aligned)
#define LDT 40

// ---- CDNA5 async global->LDS path (guarded; falls back to vector copies) ----
#if defined(__HIP_DEVICE_COMPILE__) &&                                         \
    __has_builtin(__builtin_amdgcn_global_load_async_to_lds_b128) &&           \
    __has_builtin(__builtin_amdgcn_s_wait_asynccnt)
#define USE_ASYNC 1
#else
#define USE_ASYNC 0
#endif

__device__ __forceinline__ void copy16B_lds(const unsigned short* g,
                                            unsigned short* l) {
#if USE_ASYNC
  __builtin_amdgcn_global_load_async_to_lds_b128(
      (__attribute__((address_space(1))) b128_t*)(g),
      (__attribute__((address_space(3))) b128_t*)(l), 0, 0);
#else
  *(v8u*)l = *(const v8u*)g;
#endif
}

__device__ __forceinline__ void async_fence() {
#if USE_ASYNC
  __builtin_amdgcn_s_wait_asynccnt(0);
#endif
}

__device__ __forceinline__ unsigned short f2bf(float f) {
  unsigned u = __builtin_bit_cast(unsigned, f);
  u += 0x7fffu + ((u >> 16) & 1u);          // round-to-nearest-even
  return (unsigned short)(u >> 16);
}

// A-fragment (16x32 bf16): lane half h holds K = h*16+[0..15]
__device__ __forceinline__ v16bf ldfragA(const unsigned short* p) {
  v8bf lo = *(const v8bf*)p;
  v8bf hi = *(const v8bf*)(p + 8);
  v16bf r;
#pragma unroll
  for (int i = 0; i < 8; i++) { r[i] = lo[i]; r[i + 8] = hi[i]; }
  return r;
}

// B-fragment (32x16 bf16) from column-major storage: lane half h holds
// K = h*16+[0..15] of column n=lane&15 -> one contiguous 32B chunk
__device__ __forceinline__ v16bf ldfragB(const unsigned short* p) {
  v8bf lo = *(const v8bf*)p;
  v8bf hi = *(const v8bf*)(p + 8);
  v16bf r;
#pragma unroll
  for (int i = 0; i < 8; i++) { r[i] = lo[i]; r[i + 8] = hi[i]; }
  return r;
}

// ---------------------------------------------------------------------------
// Kernel 0: bulk f32 -> bf16 conversion (vectorized, bandwidth bound)
// ---------------------------------------------------------------------------
__global__ void __launch_bounds__(256)
f32_to_bf16_kernel(const float* __restrict__ src, unsigned short* __restrict__ dst,
                   int n4) {
  int i = blockIdx.x * 256 + threadIdx.x;
  if (i >= n4) return;
  float4 v = ((const float4*)src)[i];
  v4u r = {f2bf(v.x), f2bf(v.y), f2bf(v.z), f2bf(v.w)};
  ((v4u*)dst)[i] = r;
}

// ---------------------------------------------------------------------------
// Kernels 1/3: bf16 GEMM, 128x128 block tile, 8 waves, double-buffered LDS.
//   QKV=true : C = A@B + bias scattered into per-head Q/K/V bf16 buffers
//   QKV=false: C = A@B + bias stored as f32 Out
// ---------------------------------------------------------------------------
template <int NDIM, bool QKV>
__global__ void __launch_bounds__(256)
gemm_bf16_kernel(const unsigned short* __restrict__ A,    // bf16 [M][1024]
                 const unsigned short* __restrict__ Bm,   // bf16 [1024][NDIM]
                 const float* __restrict__ bias,
                 unsigned short* __restrict__ Qb, unsigned short* __restrict__ Kb,
                 unsigned short* __restrict__ Vb, float* __restrict__ Out) {
  __shared__ unsigned short sA[2][BM * LDA];   // [m][k]
  __shared__ unsigned short sB[2][BN * LDA];   // [n][k] (B tile transposed)

  const int KD = D_MODEL;
  const int M0 = blockIdx.y * BM;
  const int N0 = blockIdx.x * BN;
  const int t = threadIdx.x;
  const int lane = t & 31;
  const int wave = t >> 5;        // 0..7
  const int wr = wave >> 1;       // M offset wr*32
  const int wc = wave & 1;        // N offset wc*64
  const int half = lane >> 4;
  const int l16 = lane & 15;

  // staging assignments
  const int arow = t >> 1, acb = (t & 1) * 16;    // A: 16 elems (32B) / thread
  const int bk = t >> 3, bnb = (t & 7) * 16;      // B: 16 elems / thread
  const unsigned short* Abase = A + (size_t)(M0 + arow) * KD + acb;

  v8f zero = {};
  v8f acc[2][4];
#pragma unroll
  for (int i = 0; i < 2; i++)
#pragma unroll
    for (int j = 0; j < 4; j++) acc[i][j] = zero;

  // ---- prologue: stage tile 0 into buffer 0
  copy16B_lds(Abase, &sA[0][arow * LDA + acb]);
  copy16B_lds(Abase + 8, &sA[0][arow * LDA + acb + 8]);
  {
    const unsigned short* bp = Bm + (size_t)bk * NDIM + N0 + bnb;
    v8u r0 = *(const v8u*)bp;
    v8u r1 = *(const v8u*)(bp + 8);
#pragma unroll
    for (int i = 0; i < 8; i++) sB[0][(bnb + i) * LDA + bk] = r0[i];
#pragma unroll
    for (int i = 0; i < 8; i++) sB[0][(bnb + 8 + i) * LDA + bk] = r1[i];
  }
  async_fence();
  __syncthreads();

  for (int k0 = 0; k0 < KD; k0 += BK) {
    const int cur = (k0 >> 5) & 1;
    const int nxt = cur ^ 1;
    const bool more = (k0 + BK) < KD;

    v8u r0, r1;
    if (more) {
      // issue next A panel async global->LDS, next B panel global->regs
      copy16B_lds(Abase + k0 + BK, &sA[nxt][arow * LDA + acb]);
      copy16B_lds(Abase + k0 + BK + 8, &sA[nxt][arow * LDA + acb + 8]);
      const unsigned short* bp = Bm + (size_t)(k0 + BK + bk) * NDIM + N0 + bnb;
      r0 = *(const v8u*)bp;
      r1 = *(const v8u*)(bp + 8);
      __builtin_prefetch(Abase + k0 + 2 * BK, 0, 1);
    }

    // ---- compute on current buffer
    v16bf a[2], b[4];
#pragma unroll
    for (int mt = 0; mt < 2; mt++)
      a[mt] = ldfragA(&sA[cur][(wr * 32 + mt * 16 + l16) * LDA + half * 16]);
#pragma unroll
    for (int nt = 0; nt < 4; nt++)
      b[nt] = ldfragB(&sB[cur][(wc * 64 + nt * 16 + l16) * LDA + half * 16]);
#pragma unroll
    for (int mt = 0; mt < 2; mt++)
#pragma unroll
      for (int nt = 0; nt < 4; nt++)
        acc[mt][nt] = __builtin_amdgcn_wmma_f32_16x16x32_bf16(
            false, a[mt], false, b[nt], (short)0, acc[mt][nt], false, false);

    if (more) {
      // scatter-transpose next B panel into the other buffer
#pragma unroll
      for (int i = 0; i < 8; i++) sB[nxt][(bnb + i) * LDA + bk] = r0[i];
#pragma unroll
      for (int i = 0; i < 8; i++) sB[nxt][(bnb + 8 + i) * LDA + bk] = r1[i];
      async_fence();
    }
    __syncthreads();
  }

  // ---- epilogue
#pragma unroll
  for (int mt = 0; mt < 2; mt++) {
#pragma unroll
    for (int nt = 0; nt < 4; nt++) {
      int c = N0 + wc * 64 + nt * 16 + l16;
      float bv = bias[c];
      if (QKV) {
        int s = c >> 10;                        // 0=Q 1=K 2=V
        int h = (c >> 6) & (NUM_HEADS - 1);
        int d = c & (D_K - 1);
        unsigned short* dst = (s == 0) ? Qb : (s == 1) ? Kb : Vb;
#pragma unroll
        for (int j = 0; j < 8; j++) {
          int m = M0 + wr * 32 + mt * 16 + j + 8 * half;
          int bb = m >> 10;
          int n = m & (SEQ - 1);
          size_t idx = (((size_t)(bb * NUM_HEADS + h) * SEQ + n) << 6) + d;
          dst[idx] = f2bf(acc[mt][nt][j] + bv);
        }
      } else {
#pragma unroll
        for (int j = 0; j < 8; j++) {
          int m = M0 + wr * 32 + mt * 16 + j + 8 * half;
          Out[(size_t)m * NDIM + c] = acc[mt][nt][j] + bv;
        }
      }
    }
  }
}

// ---------------------------------------------------------------------------
// Kernel 2: flash attention per (b,h).  4 waves x 16 query rows per block.
// ---------------------------------------------------------------------------
__global__ void __launch_bounds__(128)
attn_kernel(const unsigned short* __restrict__ Q,
            const unsigned short* __restrict__ K,
            const unsigned short* __restrict__ V,
            unsigned short* __restrict__ O) {   // attn_out bf16 [B,N,D_MODEL]
  __shared__ unsigned short vT[D_K * LDT];      // V^T chunk: [dim][key]
  __shared__ unsigned short pb[4][16 * LDT];    // per-wave P staging

  const int bh = blockIdx.y;
  const int b = bh >> 4;
  const int h = bh & (NUM_HEADS - 1);
  const int t = threadIdx.x;
  const int lane = t & 31;
  const int wave = t >> 5;
  const int half = lane >> 4;
  const int l16 = lane & 15;
  const int q0 = blockIdx.x * 64 + wave * 16;

  const size_t headoff = (size_t)(b * NUM_HEADS + h) * SEQ * D_K;
  const unsigned short* Qh = Q + headoff;
  const unsigned short* Kh = K + headoff;
  const unsigned short* Vh = V + headoff;

  v16bf qf[2];
#pragma unroll
  for (int kk = 0; kk < 2; kk++)
    qf[kk] = ldfragA(Qh + (size_t)(q0 + l16) * D_K + kk * 32 + half * 16);

  v8f zero = {};
  v8f o[4];
#pragma unroll
  for (int nt = 0; nt < 4; nt++) o[nt] = zero;
  float mrow[8], lrow[8];
#pragma unroll
  for (int j = 0; j < 8; j++) { mrow[j] = -1e30f; lrow[j] = 0.f; }

  const float scale = 0.125f;   // 1/sqrt(64)

  for (int kc = 0; kc < SEQ; kc += 32) {
    // cooperative stage of V^T chunk [64 dims][32 keys]
    {
      int key = kc + (t & 31);
      int d0 = (t >> 5) * 16;
      const unsigned short* src = Vh + (size_t)key * D_K + d0;
#pragma unroll
      for (int i = 0; i < 16; i++) vT[(d0 + i) * LDT + (t & 31)] = src[i];
    }
    __syncthreads();

    // S = Q K^T for two 16-key subtiles (K frags straight from global)
    v8f s[2];
#pragma unroll
    for (int st = 0; st < 2; st++) {
      int key0 = kc + st * 16;
      v8f sa = zero;
#pragma unroll
      for (int kk = 0; kk < 2; kk++) {
        v16bf kf = ldfragB(Kh + (size_t)(key0 + l16) * D_K + kk * 32 + half * 16);
        sa = __builtin_amdgcn_wmma_f32_16x16x32_bf16(false, qf[kk], false, kf,
                                                     (short)0, sa, false, false);
      }
      s[st] = sa;
    }

    // online softmax (row r = j + 8*half lives in one 16-lane half)
    float pv0[8], pv1[8];
#pragma unroll
    for (int j = 0; j < 8; j++) {
      float s0 = s[0][j] * scale;
      float s1 = s[1][j] * scale;
      float tm = fmaxf(s0, s1);
#pragma unroll
      for (int mk = 1; mk < 16; mk <<= 1) tm = fmaxf(tm, __shfl_xor(tm, mk, 32));
      float mnew = fmaxf(mrow[j], tm);
      float alpha = __expf(mrow[j] - mnew);
      float p0 = __expf(s0 - mnew);
      float p1 = __expf(s1 - mnew);
      float ps = p0 + p1;
#pragma unroll
      for (int mk = 1; mk < 16; mk <<= 1) ps += __shfl_xor(ps, mk, 32);
      lrow[j] = lrow[j] * alpha + ps;
      mrow[j] = mnew;
      pv0[j] = p0;
      pv1[j] = p1;
#pragma unroll
      for (int nt = 0; nt < 4; nt++) o[nt][j] *= alpha;
    }

    // repack P (C-layout) -> A-layout via per-wave LDS (in-order DS ops)
    unsigned short* P = pb[wave];
#pragma unroll
    for (int j = 0; j < 8; j++) {
      P[(j + 8 * half) * LDT + l16] = f2bf(pv0[j]);
      P[(j + 8 * half) * LDT + 16 + l16] = f2bf(pv1[j]);
    }
    v16bf pf = ldfragA(P + l16 * LDT + half * 16);

    // O += P @ V
#pragma unroll
    for (int nt = 0; nt < 4; nt++) {
      v16bf vf = ldfragB(&vT[(nt * 16 + l16) * LDT + half * 16]);
      o[nt] = __builtin_amdgcn_wmma_f32_16x16x32_bf16(false, pf, false, vf,
                                                      (short)0, o[nt], false, false);
    }
    __syncthreads();
  }

  // normalize and store bf16 attn output as [B, N, h*64 + d]
#pragma unroll
  for (int j = 0; j < 8; j++) {
    float inv = 1.f / lrow[j];
    int qrow = q0 + j + 8 * half;
    size_t rowbase = ((size_t)(b * SEQ + qrow) * D_MODEL) + h * D_K;
#pragma unroll
    for (int nt = 0; nt < 4; nt++)
      O[rowbase + nt * 16 + l16] = f2bf(o[nt][j] * inv);
  }
}

// ---------------------------------------------------------------------------
extern "C" void kernel_launch(void* const* d_in, const int* in_sizes, int n_in,
                              void* d_out, int out_size, void* d_ws, size_t ws_size,
                              hipStream_t stream) {
  (void)in_sizes; (void)n_in; (void)out_size; (void)ws_size;

  const float* x     = (const float*)d_in[0];
  const float* W_qkv = (const float*)d_in[1];
  const float* b_qkv = (const float*)d_in[2];
  const float* W_o   = (const float*)d_in[3];
  const float* b_o   = (const float*)d_in[4];
  float* out = (float*)d_out;

  // Workspace layout (bf16 elements):
  //  Q,K,V,attn_out : 4 x 8*16*1024*64     = 64 MB
  //  Xbf            : 8192*1024            = 16 MB
  //  Wqkv_bf        : 1024*3072            =  6 MB
  //  Wo_bf          : 1024*1024            =  2 MB
  const size_t HEADELEMS = (size_t)BATCH * NUM_HEADS * SEQ * D_K;
  unsigned short* qb   = (unsigned short*)d_ws;
  unsigned short* kb   = qb + HEADELEMS;
  unsigned short* vb   = kb + HEADELEMS;
  unsigned short* ao   = vb + HEADELEMS;
  unsigned short* xbf  = ao + HEADELEMS;
  unsigned short* wqkv = xbf + (size_t)BATCH * SEQ * D_MODEL;
  unsigned short* wo   = wqkv + (size_t)D_MODEL * 3 * D_MODEL;

  const int M = BATCH * SEQ;   // 8192

  int n4x = (M * D_MODEL) / 4;               // 2,097,152
  int n4q = (D_MODEL * 3 * D_MODEL) / 4;     //   786,432
  int n4o = (D_MODEL * D_MODEL) / 4;         //   262,144
  f32_to_bf16_kernel<<<(n4x + 255) / 256, 256, 0, stream>>>(x, xbf, n4x);
  f32_to_bf16_kernel<<<(n4q + 255) / 256, 256, 0, stream>>>(W_qkv, wqkv, n4q);
  f32_to_bf16_kernel<<<(n4o + 255) / 256, 256, 0, stream>>>(W_o, wo, n4o);

  dim3 g1((3 * D_MODEL) / BN, M / BM);       // 24 x 64
  gemm_bf16_kernel<3 * D_MODEL, true><<<g1, 256, 0, stream>>>(
      xbf, wqkv, b_qkv, qb, kb, vb, nullptr);

  dim3 g2(SEQ / 64, BATCH * NUM_HEADS);      // 16 x 128
  attn_kernel<<<g2, 128, 0, stream>>>(qb, kb, vb, ao);

  dim3 g3(D_MODEL / BN, M / BM);             // 8 x 64
  gemm_bf16_kernel<D_MODEL, false><<<g3, 256, 0, stream>>>(
      ao, wo, b_o, nullptr, nullptr, nullptr, out);
}
